// LSTM_49752901157090
// MI455X (gfx1250) — compile-verified, matrix operand order checked by reading, not run on hardware
//
#include <hip/hip_runtime.h>
#include <math.h>

// ---------------- problem constants ----------------
constexpr int B_  = 2048;
constexpr int T_  = 512;
constexpr int IN_ = 32;
constexpr int H_  = 64;
constexpr int G_  = 4 * H_;   // 256 gate columns
constexpr int MTILE  = 16;    // batch rows per workgroup
constexpr int NWAVES = 16;    // 16 waves * 32 lanes = 512 threads
constexpr float EPS_ = 1e-5f;

typedef _Float16 v16h __attribute__((ext_vector_type(16)));
typedef _Float16 v8h  __attribute__((ext_vector_type(8)));
typedef float    v8f  __attribute__((ext_vector_type(8)));

__device__ __forceinline__ float fast_rcp(float x) { return __builtin_amdgcn_rcpf(x); }

#if __has_builtin(__builtin_amdgcn_tanhf)
// native v_tanh_f32 (CDNA5 TRANS op)
__device__ __forceinline__ float tanhf_(float x)    { return __builtin_amdgcn_tanhf(x); }
__device__ __forceinline__ float sigmoidf_(float x) { return fmaf(0.5f, __builtin_amdgcn_tanhf(0.5f * x), 0.5f); }
#else
// sigmoid(x) = 1/(1+e^-x); tanh(x) = 1 - 2/(e^{2x}+1); limits preserved via rcp(inf)=0
__device__ __forceinline__ float sigmoidf_(float x) { return fast_rcp(1.0f + __expf(-x)); }
__device__ __forceinline__ float tanhf_(float x)    { return 1.0f - 2.0f * fast_rcp(1.0f + __expf(2.0f * x)); }
#endif

// ---------------- fused 2-layer LSTM over full time loop ----------------
// Gate columns are INTERLEAVED: physical column c = j*4 + gate (gate: 0=i,1=f,2=g,3=o)
// so a cell's four gates are one contiguous float4 in the gates buffer.
__global__ __launch_bounds__(512)
void lstm2_fused(const float* __restrict__ x,
                 const float* __restrict__ Wih0, const float* __restrict__ Whh0,
                 const float* __restrict__ bih0, const float* __restrict__ bhh0,
                 const float* __restrict__ Wih1, const float* __restrict__ Whh1,
                 const float* __restrict__ bih1, const float* __restrict__ bhh1,
                 float* __restrict__ last)
{
    __shared__ float __attribute__((aligned(16))) gates[MTILE][G_];
    __shared__ float bias0[G_], bias1[G_];
    __shared__ _Float16 __attribute__((aligned(16))) A0[MTILE][96];   // [x_t(32) | h0(64)]
    __shared__ _Float16 __attribute__((aligned(16))) A1[MTILE][128];  // [h0(64)  | h1(64)]

    const int tid  = threadIdx.x;
    const int wave = tid >> 5;
    const int lane = tid & 31;
    const int b0   = blockIdx.x * MTILE;

    if (tid < G_) { bias0[tid] = bih0[tid] + bhh0[tid]; bias1[tid] = bih1[tid] + bhh1[tid]; }

    // ---- preload B-fragments (gate weights) into registers, ISA B layout ----
    // Wave w's 16 columns map to hidden units j = 4w..4w+3, gate-interleaved.
    // lane<16 holds column cc with K 0..7,16..23 of each 32-K block; lane>=16 holds K 8..15,24..31.
    const int cc   = lane & 15;
    const int half = lane >> 4;
    const int hid  = wave * 4 + (cc >> 2);   // hidden unit j (0..63)
    const int gsel = cc & 3;                 // 0=i,1=f,2=g,3=o
    const int grow = gsel * H_ + hid;        // row in reference weight matrices
    v16h w0[3], w1[4];
#pragma unroll
    for (int kb = 0; kb < 3; ++kb) {
#pragma unroll
        for (int i = 0; i < 16; ++i) {
            int localk = (i & 7) + 8 * half + 16 * (i >> 3);
            int f = kb * 32 + localk;                       // combined feature [x | h]
            float wv = (f < IN_) ? Wih0[grow * IN_ + f] : Whh0[grow * H_ + (f - IN_)];
            w0[kb][i] = (_Float16)wv;
        }
    }
#pragma unroll
    for (int kb = 0; kb < 4; ++kb) {
#pragma unroll
        for (int i = 0; i < 16; ++i) {
            int localk = (i & 7) + 8 * half + 16 * (i >> 3);
            int f = kb * 32 + localk;                       // combined feature [h0 | h1]
            float wv = (f < H_) ? Wih1[grow * H_ + f] : Whh1[grow * H_ + (f - H_)];
            w1[kb][i] = (_Float16)wv;
        }
    }
    __syncthreads();

    // per-lane constants (after bias barrier)
    const float bv0 = bias0[grow];
    const float bv1 = bias1[grow];
    const int   scol = wave * 16 + cc;       // physical gates column this lane writes

    // ---- all recurrent state lives in registers ----
    // thread owns cells (mr, jr) and (mr+8, jr): same mapping every step.
    const int mr = tid >> 6;          // 0..7
    const int jr = tid & 63;          // 0..63
    float h0_lo = 0.f, h0_hi = 0.f, c0_lo = 0.f, c0_hi = 0.f;
    float h1_lo = 0.f, h1_hi = 0.f, c1_lo = 0.f, c1_hi = 0.f;

    // x staging mapping: thread -> (xm, xc), prefetch register
    const int xm = tid >> 5;          // 0..15
    const int xc = tid & 31;          // 0..31
    const float* xptr = x + (size_t)(b0 + xm) * (T_ * IN_) + xc;
    float xreg = xptr[0];

    const int am = lane & 15;         // A-fragment row

    for (int t = 0; t < T_; ++t) {
        // ---- stage A0 = [x_t | h0_prev] as f16 (no loops, uniform flow) ----
        A0[xm][xc]            = (_Float16)xreg;
        A0[mr][IN_ + jr]      = (_Float16)h0_lo;
        A0[mr + 8][IN_ + jr]  = (_Float16)h0_hi;
        __syncthreads();

        // prefetch next x_t (overlaps the GEMM section)
        if (t + 1 < T_) xreg = xptr[(size_t)(t + 1) * IN_];

        // ---- layer 0 gate GEMM: (16x96) x (96x16 per wave) ----
        {
            v8f acc;                                      // bias folded into acc init:
#pragma unroll
            for (int v = 0; v < 8; ++v) acc[v] = bv0;     // hides under A-fragment loads,
                                                          // removes post-WMMA adds
#pragma unroll
            for (int kb = 0; kb < 3; ++kb) {
                const v8h lo = *(const v8h*)&A0[am][kb * 32 + 8 * half];
                const v8h hi = *(const v8h*)&A0[am][kb * 32 + 8 * half + 16];
                v16h a;
#pragma unroll
                for (int i = 0; i < 8; ++i) { a[i] = lo[i]; a[i + 8] = hi[i]; }
                acc = __builtin_amdgcn_wmma_f32_16x16x32_f16(false, a, false, w0[kb],
                                                             (short)0, acc, false, false);
            }
#pragma unroll
            for (int v = 0; v < 8; ++v)
                gates[v + 8 * half][scol] = acc[v];       // C layout: M=v+8*(lane/16), N=lane%16
        }
        __syncthreads();

        // ---- layer 0 cell update (registers) + stage A1 = [h0_new | h1_prev] ----
        {
            const float4 glo = *(const float4*)&gates[mr][jr * 4];
            const float4 ghi = *(const float4*)&gates[mr + 8][jr * 4];

            float ig = sigmoidf_(glo.x);
            float fg = sigmoidf_(glo.y);
            float gg = tanhf_(glo.z);
            float og = sigmoidf_(glo.w);
            c0_lo = fg * c0_lo + ig * gg;
            h0_lo = og * tanhf_(c0_lo);

            ig = sigmoidf_(ghi.x);
            fg = sigmoidf_(ghi.y);
            gg = tanhf_(ghi.z);
            og = sigmoidf_(ghi.w);
            c0_hi = fg * c0_hi + ig * gg;
            h0_hi = og * tanhf_(c0_hi);

            A1[mr][jr]           = (_Float16)h0_lo;
            A1[mr + 8][jr]       = (_Float16)h0_hi;
            A1[mr][H_ + jr]      = (_Float16)h1_lo;
            A1[mr + 8][H_ + jr]  = (_Float16)h1_hi;
        }
        __syncthreads();

        // ---- layer 1 gate GEMM: (16x128) x (128x16 per wave) ----
        {
            v8f acc;
#pragma unroll
            for (int v = 0; v < 8; ++v) acc[v] = bv1;
#pragma unroll
            for (int kb = 0; kb < 4; ++kb) {
                const v8h lo = *(const v8h*)&A1[am][kb * 32 + 8 * half];
                const v8h hi = *(const v8h*)&A1[am][kb * 32 + 8 * half + 16];
                v16h a;
#pragma unroll
                for (int i = 0; i < 8; ++i) { a[i] = lo[i]; a[i + 8] = hi[i]; }
                acc = __builtin_amdgcn_wmma_f32_16x16x32_f16(false, a, false, w1[kb],
                                                             (short)0, acc, false, false);
            }
#pragma unroll
            for (int v = 0; v < 8; ++v)
                gates[v + 8 * half][scol] = acc[v];
        }
        __syncthreads();

        // ---- layer 1 cell update (registers only) ----
        {
            const float4 glo = *(const float4*)&gates[mr][jr * 4];
            const float4 ghi = *(const float4*)&gates[mr + 8][jr * 4];

            float ig = sigmoidf_(glo.x);
            float fg = sigmoidf_(glo.y);
            float gg = tanhf_(glo.z);
            float og = sigmoidf_(glo.w);
            c1_lo = fg * c1_lo + ig * gg;
            h1_lo = og * tanhf_(c1_lo);

            ig = sigmoidf_(ghi.x);
            fg = sigmoidf_(ghi.y);
            gg = tanhf_(ghi.z);
            og = sigmoidf_(ghi.w);
            c1_hi = fg * c1_hi + ig * gg;
            h1_hi = og * tanhf_(c1_hi);
        }
        // next iteration's A0/A1 writes don't alias gates; barrier at loop top
        // (after A0 staging) orders these gate reads vs. next wmma's gate writes.
        __syncthreads();
    }

    // final hidden state of layer 1
    last[(size_t)(b0 + mr) * H_ + jr]     = h1_lo;
    last[(size_t)(b0 + mr + 8) * H_ + jr] = h1_hi;
}

// ---------------- tail: batchnorm stats (population variance) ----------------
__global__ __launch_bounds__(256)
void bn_stats_kernel(const float* __restrict__ X, int N, int F,
                     float* __restrict__ mean, float* __restrict__ inv)
{
    int f = blockIdx.x;
    float s = 0.f, s2 = 0.f;
    for (int i = threadIdx.x; i < N; i += blockDim.x) {
        float v = X[(size_t)i * F + f];
        s += v; s2 += v * v;
    }
    __shared__ float rs[256], rs2[256];
    rs[threadIdx.x] = s; rs2[threadIdx.x] = s2;
    __syncthreads();
    for (int o = 128; o > 0; o >>= 1) {
        if (threadIdx.x < o) { rs[threadIdx.x] += rs[threadIdx.x + o]; rs2[threadIdx.x] += rs2[threadIdx.x + o]; }
        __syncthreads();
    }
    if (threadIdx.x == 0) {
        float m = rs[0] / (float)N;
        float var = rs2[0] / (float)N - m * m;
        mean[f] = m;
        inv[f]  = rsqrtf(var + EPS_);
    }
}

// ---------------- tail: bn1 + fc1 ----------------
__global__ __launch_bounds__(256)
void fc1_kernel(const float* __restrict__ last,
                const float* __restrict__ m1, const float* __restrict__ i1,
                const float* __restrict__ g1, const float* __restrict__ bb1,
                const float* __restrict__ W,  const float* __restrict__ bias,
                float* __restrict__ z)
{
    int r = threadIdx.x >> 5;          // 8 batch rows per block
    int k = threadIdx.x & 31;          // 32 output features
    int b = blockIdx.x * 8 + r;
    float acc = bias[k];
#pragma unroll 8
    for (int j = 0; j < H_; ++j) {
        float v = (last[(size_t)b * H_ + j] - m1[j]) * i1[j] * g1[j] + bb1[j];
        acc += v * W[k * H_ + j];
    }
    z[(size_t)b * 32 + k] = acc;
}

// ---------------- tail: bn2 + relu + fc2 ----------------
__global__ __launch_bounds__(256)
void fc2_kernel(const float* __restrict__ z,
                const float* __restrict__ m2, const float* __restrict__ i2,
                const float* __restrict__ g2, const float* __restrict__ bb2,
                const float* __restrict__ W,  const float* __restrict__ bias,
                float* __restrict__ out)
{
    int b = blockIdx.x * blockDim.x + threadIdx.x;
    if (b >= B_) return;
    float acc = bias[0];
#pragma unroll
    for (int k = 0; k < 32; ++k) {
        float v = (z[(size_t)b * 32 + k] - m2[k]) * i2[k] * g2[k] + bb2[k];
        v = fmaxf(v, 0.f);
        acc += v * W[k];
    }
    out[b] = acc;
}

// ---------------- launcher ----------------
extern "C" void kernel_launch(void* const* d_in, const int* in_sizes, int n_in,
                              void* d_out, int out_size, void* d_ws, size_t ws_size,
                              hipStream_t stream)
{
    const float* x    = (const float*)d_in[0];
    const float* Wih0 = (const float*)d_in[1];
    const float* Whh0 = (const float*)d_in[2];
    const float* bih0 = (const float*)d_in[3];
    const float* bhh0 = (const float*)d_in[4];
    const float* Wih1 = (const float*)d_in[5];
    const float* Whh1 = (const float*)d_in[6];
    const float* bih1 = (const float*)d_in[7];
    const float* bhh1 = (const float*)d_in[8];
    const float* bn1g = (const float*)d_in[9];
    const float* bn1b = (const float*)d_in[10];
    const float* fc1W = (const float*)d_in[11];
    const float* fc1b = (const float*)d_in[12];
    const float* bn2g = (const float*)d_in[13];
    const float* bn2b = (const float*)d_in[14];
    const float* fc2W = (const float*)d_in[15];
    const float* fc2b = (const float*)d_in[16];

    float* ws   = (float*)d_ws;
    float* last = ws;                    // 2048*64
    float* z    = last + B_ * H_;        // 2048*32
    float* m1   = z + B_ * 32;           // 64
    float* i1   = m1 + H_;               // 64
    float* m2   = i1 + H_;               // 32
    float* i2   = m2 + 32;               // 32

    lstm2_fused<<<B_ / MTILE, NWAVES * 32, 0, stream>>>(
        x, Wih0, Whh0, bih0, bhh0, Wih1, Whh1, bih1, bhh1, last);

    bn_stats_kernel<<<H_, 256, 0, stream>>>(last, B_, H_, m1, i1);
    fc1_kernel<<<B_ / 8, 256, 0, stream>>>(last, m1, i1, bn1g, bn1b, fc1W, fc1b, z);
    bn_stats_kernel<<<32, 256, 0, stream>>>(z, B_, 32, m2, i2);
    fc2_kernel<<<B_ / 256, 256, 0, stream>>>(z, m2, i2, bn2g, bn2b, fc2W, fc2b, (float*)d_out);
}
